// G2_41721312313542
// MI455X (gfx1250) — matrix operand order repaired and checked by reading, not compiled, not run on hardware
//
#include <hip/hip_runtime.h>
#include <hip/hip_bf16.h>

// ---------------------------------------------------------------------------
// GCNConv + edge abs-diff^2 scatter-mean + tanh, for MI455X (gfx1250).
// GEMM uses v_wmma_f32_16x16x32_f16 (wave32, contiguous k order for A/B).
// Edge scatter uses native global_atomic_add_f32; node arrays (25.6MB) are
// L2-resident (192MB L2), so atomics/gathers run at L2 speed.
// Workspace layout (floats): bufA[N*64] | bufB[N*64] | deg[N] | cnt[N]
//   total = 130*N*4 bytes = ~52MB for N=100000.
// ---------------------------------------------------------------------------

typedef __attribute__((ext_vector_type(16))) _Float16 v16h;
typedef __attribute__((ext_vector_type(8)))  float    v8f;

#define DFEAT 64

__device__ __forceinline__ void atomAddF(float* p, float v) {
    (void)unsafeAtomicAdd(p, v);   // lowers to global_atomic_add_f32 on gfx1250
}

// ---- init: deg = 1 (self loop), cnt = 0 -----------------------------------
__global__ void k_init(float* __restrict__ deg, float* __restrict__ cnt, int n) {
    int i = blockIdx.x * blockDim.x + threadIdx.x;
    if (i < n) { deg[i] = 1.0f; cnt[i] = 0.0f; }
}

// ---- degree accumulation over col -----------------------------------------
__global__ void k_deg(const int* __restrict__ col, float* __restrict__ deg, int E) {
    int e = blockIdx.x * blockDim.x + threadIdx.x;
    if (e < E) atomAddF(&deg[col[e]], 1.0f);
}

// ---- deg -> dinv = rsqrt(deg) (deg >= 1 always) ---------------------------
__global__ void k_rsqrt(float* __restrict__ deg, int n) {
    int i = blockIdx.x * blockDim.x + threadIdx.x;
    if (i < n) deg[i] = rsqrtf(deg[i]);
}

// ---- H = X @ W via WMMA f32_16x16x32_f16 ----------------------------------
// One wave handles a 16-row x 64-col output tile; 8 waves per block.
// W (64x64 f32) is transposed+converted to f16 in LDS once per block.
__global__ void k_gemm(const float* __restrict__ X, const float* __restrict__ W,
                       float* __restrict__ H, int n) {
    __shared__ _Float16 Wt[DFEAT * DFEAT];   // Wt[col][k] = W[k][col]
    for (int idx = threadIdx.x; idx < DFEAT * DFEAT; idx += blockDim.x) {
        int k = idx >> 6, c = idx & 63;
        Wt[c * DFEAT + k] = (_Float16)W[idx];
    }
    __syncthreads();

    const int lane  = threadIdx.x & 31;
    const int wave  = threadIdx.x >> 5;
    const int tile  = blockIdx.x * (blockDim.x >> 5) + wave;
    const int ntile = (n + 15) >> 4;
    if (tile >= ntile) return;               // wave-uniform branch (EXEC stays full)

    const int row0 = tile << 4;
    const int m    = lane & 15;              // M (for A) / N (for B,C)
    const int hi   = (lane >> 4) & 1;        // lane half selects K/M split
    const int row  = row0 + m;

    v8f acc[4];
    #pragma unroll
    for (int t = 0; t < 4; ++t) acc[t] = (v8f){0.f,0.f,0.f,0.f,0.f,0.f,0.f,0.f};

    #pragma unroll
    for (int kc = 0; kc < 2; ++kc) {         // K = 64 in two chunks of 32
        const int k0 = kc << 5;
        // A operand: contiguous k order (this code's choice). element e -> K = e + hi*16
        v16h a;
        #pragma unroll
        for (int e = 0; e < 16; ++e) a[e] = (_Float16)0.0f;
        if (row < n) {
            const float* xr = X + (size_t)row * DFEAT + k0 + hi * 16;
            #pragma unroll
            for (int e = 0; e < 16; ++e) {
                int K = e;
                a[e] = (_Float16)xr[K];
            }
        }
        const int kb = k0 + hi * 16;         // B: lanes 0-15 -> K=0..15, hi -> K=16..31
        #pragma unroll
        for (int nc = 0; nc < 4; ++nc) {
            const v16h b = *(const v16h*)&Wt[(size_t)(nc * 16 + m) * DFEAT + kb];
            acc[nc] = __builtin_amdgcn_wmma_f32_16x16x32_f16(
                false, a, false, b, (short)0, acc[nc], false, false);
        }
    }

    // C/D layout: VGPR r -> M = r + hi*8, N = lane%16
    #pragma unroll
    for (int nc = 0; nc < 4; ++nc) {
        #pragma unroll
        for (int r = 0; r < 8; ++r) {
            int rr = row0 + r + hi * 8;
            if (rr < n) H[(size_t)rr * DFEAT + nc * 16 + m] = acc[nc][r];
        }
    }
}

// ---- agg = h * dinv^2 (self-loop term, full overwrite of bufB) ------------
__global__ void k_selfinit(const float* __restrict__ H, const float* __restrict__ dinv,
                           float* __restrict__ agg, int n) {
    int t = blockIdx.x * blockDim.x + threadIdx.x;   // n*16 float4 chunks
    if (t >= n * 16) return;
    int i = t >> 4;
    float s = dinv[i] * dinv[i];
    float4 v = ((const float4*)H)[t];
    float4 o; o.x = v.x * s; o.y = v.y * s; o.z = v.z * s; o.w = v.w * s;
    ((float4*)agg)[t] = o;
}

// ---- edge aggregation: agg[col] += h[row] * dinv[row]*dinv[col] -----------
__global__ void k_edgeagg(const float* __restrict__ H, const float* __restrict__ dinv,
                          const int* __restrict__ row, const int* __restrict__ col,
                          float* __restrict__ agg, int E) {
    int t = blockIdx.x * blockDim.x + threadIdx.x;   // E*16 threads, float4 each
    if (t >= E * 16) return;
    int e = t >> 4, j = t & 15;
    int r = row[e], c = col[e];
    float nrm = dinv[r] * dinv[c];
    float4 v = ((const float4*)H)[(size_t)r * 16 + j];
    float* dst = agg + (size_t)c * DFEAT + j * 4;
    atomAddF(dst + 0, v.x * nrm);
    atomAddF(dst + 1, v.y * nrm);
    atomAddF(dst + 2, v.z * nrm);
    atomAddF(dst + 3, v.w * nrm);
}

// ---- Xc = relu(agg + b); also zero agg (reused as s next) -----------------
__global__ void k_bias_relu(float* __restrict__ agg, const float* __restrict__ bias,
                            float* __restrict__ Xc, int n) {
    int t = blockIdx.x * blockDim.x + threadIdx.x;
    if (t >= n * 16) return;
    int j = t & 15;
    float4 v = ((const float4*)agg)[t];
    float4 b = ((const float4*)bias)[j];
    float4 o;
    o.x = fmaxf(v.x + b.x, 0.0f);
    o.y = fmaxf(v.y + b.y, 0.0f);
    o.z = fmaxf(v.z + b.z, 0.0f);
    o.w = fmaxf(v.w + b.w, 0.0f);
    ((float4*)Xc)[t] = o;
    float4 z; z.x = 0.f; z.y = 0.f; z.z = 0.f; z.w = 0.f;
    ((float4*)agg)[t] = z;                    // s buffer zeroed for next phase
}

// ---- per-edge |Xc[src]-Xc[dst]|^2 scatter-add over src, count edges -------
__global__ void k_edgediff(const float* __restrict__ Xc,
                           const int* __restrict__ row, const int* __restrict__ col,
                           float* __restrict__ s, float* __restrict__ cnt, int E) {
    int t = blockIdx.x * blockDim.x + threadIdx.x;
    if (t >= E * 16) return;
    int e = t >> 4, j = t & 15;
    int sr = row[e], ds = col[e];
    float4 a = ((const float4*)Xc)[(size_t)sr * 16 + j];
    float4 b = ((const float4*)Xc)[(size_t)ds * 16 + j];
    float dx = a.x - b.x, dy = a.y - b.y, dz = a.z - b.z, dw = a.w - b.w;
    float* dst = s + (size_t)sr * DFEAT + j * 4;
    atomAddF(dst + 0, dx * dx);               // P = 2 -> |d|^2 = d*d
    atomAddF(dst + 1, dy * dy);
    atomAddF(dst + 2, dz * dz);
    atomAddF(dst + 3, dw * dw);
    if (j == 0) atomAddF(&cnt[sr], 1.0f);
}

// ---- out = tanh(s / max(cnt,1)) -------------------------------------------
__global__ void k_final(const float* __restrict__ s, const float* __restrict__ cnt,
                        float* __restrict__ out, int n) {
    int t = blockIdx.x * blockDim.x + threadIdx.x;
    if (t >= n * 16) return;
    int i = t >> 4;
    float inv = 1.0f / fmaxf(cnt[i], 1.0f);
    float4 v = ((const float4*)s)[t];
    float4 o;
    o.x = tanhf(v.x * inv);
    o.y = tanhf(v.y * inv);
    o.z = tanhf(v.z * inv);
    o.w = tanhf(v.w * inv);
    ((float4*)out)[t] = o;
}

extern "C" void kernel_launch(void* const* d_in, const int* in_sizes, int n_in,
                              void* d_out, int out_size, void* d_ws, size_t ws_size,
                              hipStream_t stream) {
    const float* X    = (const float*)d_in[0];   // [N,64] f32
    const float* W    = (const float*)d_in[1];   // [64,64] f32
    const float* bias = (const float*)d_in[2];   // [64] f32
    const int*   ei   = (const int*)d_in[3];     // [2,E] i32

    const int n = in_sizes[0] / DFEAT;
    const int E = in_sizes[3] / 2;
    const int* row = ei;          // edge_index[0] (src / gather for conv)
    const int* col = ei + E;      // edge_index[1] (dst / scatter target)

    float* ws   = (float*)d_ws;
    float* bufA = ws;                              // h, later Xc
    float* bufB = ws + (size_t)n * DFEAT;          // agg, later s
    float* deg  = ws + (size_t)2 * n * DFEAT;      // deg -> dinv
    float* cnt  = deg + n;
    float* out  = (float*)d_out;

    const int TPB = 256;
    const int nb_n    = (n + TPB - 1) / TPB;
    const int nb_e    = (E + TPB - 1) / TPB;
    const int nb_nf4  = (n * 16 + TPB - 1) / TPB;
    const int nb_ef4  = (int)(((size_t)E * 16 + TPB - 1) / TPB);
    const int ntiles  = (n + 15) / 16;
    const int nb_gemm = (ntiles + 7) / 8;          // 8 waves (wave32) per block

    k_init     <<<nb_n,    TPB, 0, stream>>>(deg, cnt, n);
    k_deg      <<<nb_e,    TPB, 0, stream>>>(col, deg, E);
    k_rsqrt    <<<nb_n,    TPB, 0, stream>>>(deg, n);
    k_gemm     <<<nb_gemm, TPB, 0, stream>>>(X, W, bufA, n);
    k_selfinit <<<nb_nf4,  TPB, 0, stream>>>(bufA, deg, bufB, n);
    k_edgeagg  <<<nb_ef4,  TPB, 0, stream>>>(bufA, deg, row, col, bufB, E);
    k_bias_relu<<<nb_nf4,  TPB, 0, stream>>>(bufB, bias, bufA, n);
    k_edgediff <<<nb_ef4,  TPB, 0, stream>>>(bufA, row, col, bufB, cnt, E);
    k_final    <<<nb_nf4,  TPB, 0, stream>>>(bufB, cnt, out, n);
}
